// SambaBlock_87883620810779
// MI455X (gfx1250) — compile-verified, matrix operand order checked by reading, not run on hardware
//
#include <hip/hip_runtime.h>
#include <hip/hip_bf16.h>
#include <math.h>

// ---------------- problem constants ----------------
#define BATCH   8
#define LTOK    4096
#define DIM     384
#define D_STATE 64
#define D_CONV  4
#define D_IN    768          // EXPAND*DIM
#define DT_RANK 24
#define HID     1536         // 4*DIM
#define NTOK    (BATCH*LTOK) // 32768
#define XPN     (DT_RANK + 2*D_STATE) // 152
#define GRID_H  64
#define GRID_W  64

typedef __attribute__((ext_vector_type(16))) __bf16 v16bf;
typedef __attribute__((ext_vector_type(8)))  float  v8f;

// ============================================================
// gfx1250 async global->LDS copy (ASYNCcnt-tracked), 16B/lane
// ============================================================
__device__ __forceinline__ void async_copy16(unsigned lds_addr, const __bf16* gsrc) {
    asm volatile("global_load_async_to_lds_b128 %0, %1, off"
                 :: "v"(lds_addr), "v"(gsrc) : "memory");
}
__device__ __forceinline__ void wait_async0() {
    asm volatile("s_wait_asynccnt 0x0" ::: "memory");
}

// ============================================================
// fp32 -> bf16 conversion (weights)
// ============================================================
__global__ void cvt_f32_bf16(const float* __restrict__ s, __bf16* __restrict__ d, int n) {
    int i = blockIdx.x * blockDim.x + threadIdx.x;
    if (i < n) d[i] = (__bf16)s[i];
}

// ============================================================
// LayerNorm over last dim (384), one wave per token, bf16 out
// ============================================================
__global__ void __launch_bounds__(256)
ln_bf16(const float* __restrict__ X, const float* __restrict__ gam,
        const float* __restrict__ bet, __bf16* __restrict__ out, int ntok) {
    int lane = threadIdx.x & 31;
    int wave = threadIdx.x >> 5;
    int tok  = blockIdx.x * 8 + wave;
    if (tok >= ntok) return;
    const float* row = X + (size_t)tok * DIM;
    float v[DIM / 32];
    float s = 0.f;
#pragma unroll
    for (int i = 0; i < DIM / 32; i++) { v[i] = row[lane + i * 32]; s += v[i]; }
#pragma unroll
    for (int o = 16; o > 0; o >>= 1) s += __shfl_xor(s, o, 32);
    float mu = s * (1.f / DIM);
    float q = 0.f;
#pragma unroll
    for (int i = 0; i < DIM / 32; i++) { float d = v[i] - mu; q += d * d; }
#pragma unroll
    for (int o = 16; o > 0; o >>= 1) q += __shfl_xor(q, o, 32);
    float rs = rsqrtf(q * (1.f / DIM) + 1e-5f);
    __bf16* orow = out + (size_t)tok * DIM;
#pragma unroll
    for (int i = 0; i < DIM / 32; i++) {
        int c = lane + i * 32;
        orow[c] = (__bf16)((v[i] - mu) * rs * gam[c] + bet[c]);
    }
}

// ============================================================
// WMMA bf16 GEMM:  C[M,N] = A[M,K] * W[N,K]^T  (+bias)(+resid)
// Block = 8 waves = 128(M) x 64(N) tile. The shared 64x32 W tile
// is staged to LDS with async global->LDS copies, double-buffered
// (stage of k+32 overlaps WMMA on k). A streams from global with
// prefetch. B fragment (KxN) of W^T is read from LDS with the
// standard A-fragment lane pattern applied to W rows. All 4 B
// fragments are loaded before the 4 WMMAs so the ds_loads clause
// under one wait and the WMMAs issue back-to-back.
// ============================================================
__device__ __forceinline__ v16bf gmem_frag(const __bf16* __restrict__ mat, int rowBase,
                                           int rowMax, int ld, int k0, int lane) {
    int r = rowBase + (lane & 15);
    r = r < rowMax ? r : rowMax - 1;
    int klo = k0 + ((lane >> 4) << 3);
    const __bf16* p = mat + (size_t)r * ld + klo;
    union { v16bf v; uint4 q[2]; } u;
    u.q[0] = *(const uint4*)(p);
    u.q[1] = *(const uint4*)(p + 16);
    return u.v;
}

__device__ __forceinline__ v16bf lds_frag(const char* __restrict__ sbuf, int t, int lane) {
    // LDS tile: 64 rows x 32 bf16, row stride 64B
    int off = (t * 16 + (lane & 15)) * 64 + ((lane >> 4) << 4);
    union { v16bf v; uint4 q[2]; } u;
    u.q[0] = *(const uint4*)(sbuf + off);
    u.q[1] = *(const uint4*)(sbuf + off + 32);
    return u.v;
}

__device__ __forceinline__ void stage_w_async(const __bf16* __restrict__ W, int n0, int N,
                                              int K, int k0, unsigned lds_base, int tid) {
    int row = tid >> 2;                 // 0..63
    int seg = tid & 3;                  // 4 x 16B per row
    int r = n0 + row; r = r < N ? r : N - 1;
    async_copy16(lds_base + row * 64 + seg * 16,
                 W + (size_t)r * K + k0 + seg * 8);
}

__global__ void __launch_bounds__(256)
gemm_bf16_wmma(const __bf16* __restrict__ A, const __bf16* __restrict__ W,
               int M, int N, int K,
               const float* __restrict__ bias, const float* __restrict__ resid,
               float* __restrict__ outF, __bf16* __restrict__ outB) {
    __shared__ __align__(16) char smem[2][64 * 64];   // 2 x 4KB double buffer
    int tid  = threadIdx.x;
    int lane = tid & 31;
    int wave = tid >> 5;
    int m0 = blockIdx.y * 128 + wave * 16;   // M always a multiple of 128 here
    int n0 = blockIdx.x * 64;
    unsigned lbase[2] = { (unsigned)(size_t)&smem[0][0],
                          (unsigned)(size_t)&smem[1][0] };

    // prologue: stage k=0 tile into buffer 0
    stage_w_async(W, n0, N, K, 0, lbase[0], tid);

    v8f acc[4] = {v8f{}, v8f{}, v8f{}, v8f{}};
    int cur = 0;
    for (int k0 = 0; k0 < K; k0 += 32) {
        wait_async0();          // own stage of buf[cur] complete
        __syncthreads();        // whole tile ready; prev reads of buf[cur^1] done
        if (k0 + 32 < K) {
            stage_w_async(W, n0, N, K, k0 + 32, lbase[cur ^ 1], tid);  // overlap
            __builtin_prefetch(A + (size_t)(m0 + (lane & 15)) * K + k0 + 32, 0, 1);
        }
        v16bf a = gmem_frag(A, m0, M, K, k0, lane);
        const char* sb = smem[cur];
        // phase 1: clause all 8 ds_load_b128 under one dscnt wait
        v16bf b[4];
#pragma unroll
        for (int t = 0; t < 4; t++) b[t] = lds_frag(sb, t, lane);
        // phase 2: back-to-back WMMAs (independent accumulators)
#pragma unroll
        for (int t = 0; t < 4; t++)
            acc[t] = __builtin_amdgcn_wmma_f32_16x16x32_bf16(
                false, a, false, b[t], (short)0, acc[t], false, false);
        cur ^= 1;
    }
    // epilogue: D layout -> VGPR r, lanes 0-15: M=r, lanes 16-31: M=r+8
#pragma unroll
    for (int t = 0; t < 4; t++) {
        int n = n0 + t * 16 + (lane & 15);
        if (n >= N) continue;
        float bv = bias ? bias[n] : 0.f;
#pragma unroll
        for (int r = 0; r < 8; r++) {
            int m = m0 + r + ((lane >> 4) << 3);
            size_t idx = (size_t)m * N + n;
            float v = acc[t][r] + bv;
            if (resid) v += resid[idx];
            if (outF) outF[idx] = v;
            if (outB) outB[idx] = (__bf16)v;
        }
    }
}

// ============================================================
// causal depthwise conv1d (k=4) + bias + SiLU : xz[:, :768] -> xc
// ============================================================
__global__ void __launch_bounds__(256)
conv1d_silu(const __bf16* __restrict__ xz, const float* __restrict__ cw,
            const float* __restrict__ cb, __bf16* __restrict__ xc) {
    size_t i = (size_t)blockIdx.x * blockDim.x + threadIdx.x;
    if (i >= (size_t)NTOK * D_IN) return;
    int d = (int)(i % D_IN);
    int tok = (int)(i / D_IN);
    int b = tok / LTOK, l = tok % LTOK;
    float acc = cb[d];
#pragma unroll
    for (int j = 0; j < D_CONV; j++) {
        int li = l - (D_CONV - 1) + j;
        if (li >= 0)
            acc += (float)xz[((size_t)(b * LTOK + li)) * (2 * D_IN) + d] * cw[d * D_CONV + j];
    }
    float s = acc / (1.f + __expf(-acc)); // SiLU
    xc[i] = (__bf16)s;
}

// ============================================================
// dt = softplus(xdb[:, :24] @ W_dt^T + b_dt)   (K=24, plain VALU)
// ============================================================
__global__ void __launch_bounds__(256)
dt_kernel(const float* __restrict__ xdb, const float* __restrict__ Wdt,
          const float* __restrict__ bdt, float* __restrict__ dt) {
    size_t i = (size_t)blockIdx.x * blockDim.x + threadIdx.x;
    if (i >= (size_t)NTOK * D_IN) return;
    int d = (int)(i % D_IN);
    int tok = (int)(i / D_IN);
    const float* xr = xdb + (size_t)tok * XPN;
    float acc = bdt[d];
#pragma unroll
    for (int r = 0; r < DT_RANK; r++) acc += xr[r] * Wdt[d * DT_RANK + r];
    dt[i] = (acc > 20.f) ? acc : log1pf(__expf(acc)); // softplus
}

// ============================================================
// selective scan: one wave = 32 channels of one batch;
// lane holds h[64] + A[64] in registers, 4096 serial steps.
// ============================================================
__global__ void __launch_bounds__(32)
mamba_scan(const float* __restrict__ dt, const float* __restrict__ xdb,
           const __bf16* __restrict__ xc, const float* __restrict__ A_log,
           float* __restrict__ y) {
    int lane = threadIdx.x;
    int cb = blockIdx.x % (D_IN / 32);
    int b  = blockIdx.x / (D_IN / 32);
    int d  = cb * 32 + lane;
    float A[D_STATE], h[D_STATE];
#pragma unroll
    for (int s = 0; s < D_STATE; s++) {
        A[s] = -__expf(A_log[(size_t)d * D_STATE + s]);
        h[s] = 0.f;
    }
    for (int l = 0; l < LTOK; l++) {
        size_t tok = (size_t)b * LTOK + l;
        float dtv = dt[tok * D_IN + d];
        float xv  = (float)xc[tok * D_IN + d];
        float dx  = dtv * xv;
        const float* Bp = xdb + tok * XPN + DT_RANK;
        const float* Cp = Bp + D_STATE;
        float acc = 0.f;
#pragma unroll
        for (int s = 0; s < D_STATE; s++) {
            h[s] = h[s] * __expf(dtv * A[s]) + dx * Bp[s];
            acc += h[s] * Cp[s];
        }
        y[tok * D_IN + d] = acc;
    }
}

// ============================================================
// gating: ybf = (y + xc*Dp) * silu(z) ; z = xz[:, 768:]
// ============================================================
__global__ void __launch_bounds__(256)
gate_kernel(const float* __restrict__ y, const __bf16* __restrict__ xc,
            const float* __restrict__ Dp, const __bf16* __restrict__ xz,
            __bf16* __restrict__ ybf) {
    size_t i = (size_t)blockIdx.x * blockDim.x + threadIdx.x;
    if (i >= (size_t)NTOK * D_IN) return;
    int d = (int)(i % D_IN);
    size_t tok = i / D_IN;
    float zv = (float)xz[tok * (2 * D_IN) + D_IN + d];
    float yv = y[i] + (float)xc[i] * Dp[d];
    ybf[i] = (__bf16)(yv * (zv / (1.f + __expf(-zv))));
}

// ============================================================
// depthwise 3x3 conv over 64x64 grid + bias + exact GELU
// ============================================================
__global__ void __launch_bounds__(256)
dwconv_gelu(const __bf16* __restrict__ f, const float* __restrict__ w,
            const float* __restrict__ bia, __bf16* __restrict__ g) {
    size_t i = (size_t)blockIdx.x * blockDim.x + threadIdx.x;
    if (i >= (size_t)NTOK * HID) return;
    int c = (int)(i % HID);
    int tok = (int)(i / HID);
    int b = tok / LTOK, l = tok % LTOK;
    int yy = l / GRID_W, xx = l % GRID_W;
    float acc = bia[c];
#pragma unroll
    for (int dy = 0; dy < 3; dy++) {
#pragma unroll
        for (int dx = 0; dx < 3; dx++) {
            int iy = yy + dy - 1, ix = xx + dx - 1;
            if (iy >= 0 && iy < GRID_H && ix >= 0 && ix < GRID_W)
                acc += (float)f[((size_t)(b * LTOK + iy * GRID_W + ix)) * HID + c] *
                       w[c * 9 + dy * 3 + dx];
        }
    }
    g[i] = (__bf16)(0.5f * acc * (1.f + erff(acc * 0.70710678118654752f)));
}

// ============================================================
// host-side orchestration
// ============================================================
extern "C" void kernel_launch(void* const* d_in, const int* in_sizes, int n_in,
                              void* d_out, int out_size, void* d_ws, size_t ws_size,
                              hipStream_t stream) {
    const float* x     = (const float*)d_in[0];
    const float* g1    = (const float*)d_in[3];
    const float* be1   = (const float*)d_in[4];
    const float* W_in  = (const float*)d_in[5];
    const float* convw = (const float*)d_in[6];
    const float* convb = (const float*)d_in[7];
    const float* W_xp  = (const float*)d_in[8];
    const float* W_dt  = (const float*)d_in[9];
    const float* b_dt  = (const float*)d_in[10];
    const float* A_log = (const float*)d_in[11];
    const float* Dp    = (const float*)d_in[12];
    const float* W_out = (const float*)d_in[13];
    const float* g2    = (const float*)d_in[14];
    const float* be2   = (const float*)d_in[15];
    const float* W1    = (const float*)d_in[16];
    const float* b1    = (const float*)d_in[17];
    const float* dw_w  = (const float*)d_in[18];
    const float* dw_b  = (const float*)d_in[19];
    const float* W2    = (const float*)d_in[20];
    const float* b2    = (const float*)d_in[21];
    float* out = (float*)d_out;

    // ---- workspace layout (bytes, 256-aligned), aliased for peak ~502MB ----
    char* ws = (char*)d_ws;
    const size_t OFF_WIN  = 0;                                 // 1536*384 bf16
    const size_t OFF_WXP  = OFF_WIN  + (size_t)HID*DIM*2;      //  152*768 bf16
    const size_t OFF_WOUT = OFF_WXP  + (size_t)XPN*D_IN*2;     //  384*768 bf16
    const size_t OFF_W1   = OFF_WOUT + (size_t)DIM*D_IN*2;     // 1536*384 bf16
    const size_t OFF_W2   = OFF_W1   + (size_t)HID*DIM*2;      //  384*1536 bf16
    const size_t OFF_XN   = OFF_W2   + (size_t)DIM*HID*2;      // 32768*384 bf16 (reused: xn2)
    const size_t OFF_XZ   = OFF_XN   + (size_t)NTOK*DIM*2;     // 32768*1536 bf16 (reused: f)
    const size_t OFF_XC   = OFF_XZ   + (size_t)NTOK*2*D_IN*2;  // 32768*768 bf16
    const size_t OFF_XDB  = OFF_XC   + (size_t)NTOK*D_IN*2;    // 32768*152 f32
    const size_t OFF_DT   = OFF_XDB  + (size_t)NTOK*XPN*4;     // 32768*768 f32 (reused: g bf16)
    const size_t OFF_Y    = OFF_DT   + (size_t)NTOK*D_IN*4;    // 32768*768 f32
    const size_t OFF_YBF  = OFF_Y    + (size_t)NTOK*D_IN*4;    // 32768*768 bf16
    const size_t OFF_X2   = OFF_YBF  + (size_t)NTOK*D_IN*2;    // 32768*384 f32

    __bf16* wbf_in  = (__bf16*)(ws + OFF_WIN);
    __bf16* wbf_xp  = (__bf16*)(ws + OFF_WXP);
    __bf16* wbf_out = (__bf16*)(ws + OFF_WOUT);
    __bf16* wbf_w1  = (__bf16*)(ws + OFF_W1);
    __bf16* wbf_w2  = (__bf16*)(ws + OFF_W2);
    __bf16* xn      = (__bf16*)(ws + OFF_XN);
    __bf16* xz      = (__bf16*)(ws + OFF_XZ);
    __bf16* xc      = (__bf16*)(ws + OFF_XC);
    float*  xdb     = (float*) (ws + OFF_XDB);
    float*  dt      = (float*) (ws + OFF_DT);
    float*  y       = (float*) (ws + OFF_Y);
    __bf16* ybf     = (__bf16*)(ws + OFF_YBF);
    float*  x2      = (float*) (ws + OFF_X2);
    __bf16* f   = xz;             // alias (z consumed before W1 GEMM)
    __bf16* g   = (__bf16*)dt;    // alias (dt consumed; same byte size)
    __bf16* xn2 = xn;             // alias

    const int T = 256;
    auto blks = [](size_t n, int t) { return (unsigned)((n + t - 1) / t); };

    // 1) weight fp32 -> bf16
    cvt_f32_bf16<<<blks((size_t)HID*DIM, T), T, 0, stream>>>(W_in,  wbf_in,  HID*DIM);
    cvt_f32_bf16<<<blks((size_t)XPN*D_IN, T), T, 0, stream>>>(W_xp,  wbf_xp,  XPN*D_IN);
    cvt_f32_bf16<<<blks((size_t)DIM*D_IN, T), T, 0, stream>>>(W_out, wbf_out, DIM*D_IN);
    cvt_f32_bf16<<<blks((size_t)HID*DIM, T), T, 0, stream>>>(W1,    wbf_w1,  HID*DIM);
    cvt_f32_bf16<<<blks((size_t)DIM*HID, T), T, 0, stream>>>(W2,    wbf_w2,  DIM*HID);

    // 2) LN1: x -> xn (bf16)
    ln_bf16<<<NTOK / 8, T, 0, stream>>>(x, g1, be1, xn, NTOK);

    // 3) xz = xn @ W_in^T   [32768,384]x[1536,384] -> bf16
    gemm_bf16_wmma<<<dim3((2*D_IN + 63) / 64, NTOK / 128), T, 0, stream>>>(
        xn, wbf_in, NTOK, 2*D_IN, DIM, nullptr, nullptr, nullptr, xz);

    // 4) causal conv1d + SiLU -> xc
    conv1d_silu<<<blks((size_t)NTOK * D_IN, T), T, 0, stream>>>(xz, convw, convb, xc);

    // 5) xdb = xc @ W_xproj^T  [.,768]x[152,768] -> f32 (N=152 edge handled)
    gemm_bf16_wmma<<<dim3((XPN + 63) / 64, NTOK / 128), T, 0, stream>>>(
        xc, wbf_xp, NTOK, XPN, D_IN, nullptr, nullptr, xdb, nullptr);

    // 6) dt = softplus(xdb[:, :24] @ W_dt^T + b_dt)
    dt_kernel<<<blks((size_t)NTOK * D_IN, T), T, 0, stream>>>(xdb, W_dt, b_dt, dt);

    // 7) selective scan (serial over 4096 tokens)
    mamba_scan<<<BATCH * (D_IN / 32), 32, 0, stream>>>(dt, xdb, xc, A_log, y);

    // 8) ybf = (y + xc*Dp) * silu(z)
    gate_kernel<<<blks((size_t)NTOK * D_IN, T), T, 0, stream>>>(y, xc, Dp, xz, ybf);

    // 9) x2 = x + ybf @ W_out^T
    gemm_bf16_wmma<<<dim3((DIM + 63) / 64, NTOK / 128), T, 0, stream>>>(
        ybf, wbf_out, NTOK, DIM, D_IN, nullptr, x, x2, nullptr);

    // 10) LN2: x2 -> xn2
    ln_bf16<<<NTOK / 8, T, 0, stream>>>(x2, g2, be2, xn2, NTOK);

    // 11) f = xn2 @ W1^T + b1 -> bf16
    gemm_bf16_wmma<<<dim3((HID + 63) / 64, NTOK / 128), T, 0, stream>>>(
        xn2, wbf_w1, NTOK, HID, DIM, b1, nullptr, nullptr, f);

    // 12) depthwise 3x3 + bias + GELU -> g
    dwconv_gelu<<<blks((size_t)NTOK * HID, T), T, 0, stream>>>(f, dw_w, dw_b, g);

    // 13) out = x2 + g @ W2^T + b2
    gemm_bf16_wmma<<<dim3((DIM + 63) / 64, NTOK / 128), T, 0, stream>>>(
        g, wbf_w2, NTOK, DIM, HID, b2, x2, out, nullptr);
}
